// GAT_Edge_feat_4492535792527
// MI455X (gfx1250) — matrix operand order repaired and checked by reading, not compiled, not run on hardware
//
#include <hip/hip_runtime.h>
#include <math.h>

typedef __attribute__((ext_vector_type(16))) _Float16 v16h;
typedef __attribute__((ext_vector_type(8)))  float    v8f;

// ---------------------------------------------------------------- WMMA helpers
static __device__ __forceinline__ v8f wmma_f16(v16h a, v16h b, v8f c) {
  // D = A(16x32 f16) * B(32x16 f16) + C(16x16 f32)
  return __builtin_amdgcn_wmma_f32_16x16x32_f16(false, a, false, b, (short)0, c,
                                                false, false);
}

// A-matrix fragment (16-bit, 16x32 MxK): lanes 0-15 -> M=lane, K=0..15;
// lanes 16-31 -> M=lane-16, K=16..31.
static __device__ __forceinline__ v16h build_a(const _Float16* t, int ldk,
                                               int lane, int k0) {
  const int m  = lane & 15;
  const int kb = (lane & 16) ? 16 : 0;
  const _Float16* r = t + m * ldk + k0 + kb;
  v16h a;
#pragma unroll
  for (int j = 0; j < 8; ++j) { a[j] = r[j]; a[j + 8] = r[8 + j]; }
  return a;
}

// B-matrix fragment (16-bit, 32x16 KxN): lanes 0-15 -> N=lane, K=0..15;
// lanes 16-31 -> N=lane-16, K=16..31.
static __device__ __forceinline__ v16h build_b(const _Float16* w, int ldn,
                                               int lane, int k0, int nh) {
  const int col = nh * 16 + (lane & 15);
  const int kb  = (lane & 16) ? 16 : 0;
  v16h b;
#pragma unroll
  for (int j = 0; j < 16; ++j) b[j] = w[(k0 + kb + j) * ldn + col];
  return b;
}

// ------------------------------------------------- ordered float <-> uint (max)
static __device__ __forceinline__ unsigned encf(float f) {
  unsigned u = __float_as_uint(f);
  return (u & 0x80000000u) ? ~u : (u | 0x80000000u);
}
static __device__ __forceinline__ float decf(unsigned u) {
  return __uint_as_float((u & 0x80000000u) ? (u & 0x7FFFFFFFu) : ~u);
}

// ------------------------------------------------------------------ edge fetch
static __device__ __forceinline__ void edge_fetch(long long e, const int* ep,
                                                  const int* es, const int* ev,
                                                  long long E, float meanEA,
                                                  int* src, int* dst, float* ea) {
  if (e < E) {
    *src = ep[e]; *dst = ep[E + e]; *ea = 1.0f;
  } else if (e < 2 * E) {
    long long i = e - E;     *src = es[i]; *dst = es[E + i]; *ea = 2.0f;
  } else if (e < 3 * E) {
    long long i = e - 2 * E; *src = ev[i]; *dst = ev[E + i]; *ea = 3.0f;
  } else {
    int i = (int)(e - 3 * E); *src = i; *dst = i; *ea = meanEA;  // self-loop
  }
}

// ---------------------------------------------------------------- init buffers
__global__ void init_kernel(unsigned* __restrict__ menc, float* __restrict__ denom,
                            float* __restrict__ h, int N) {
  long long i = (long long)blockIdx.x * blockDim.x + threadIdx.x;
  if (i < (long long)N * 32) h[i] = 0.0f;
  if (i < N) {
    menc[i]  = encf(-__builtin_huge_valf());
    denom[i] = 0.0f;
  }
}

// --------------------------------------------- xl = x@Wl+bl ; xr = x@Wr+br (WMMA)
__global__ __launch_bounds__(128) void node_transform_kernel(
    const float* __restrict__ x,
    const float* __restrict__ Wl, const float* __restrict__ bl,
    const float* __restrict__ Wr, const float* __restrict__ br,
    float* __restrict__ xl, float* __restrict__ xr, int N) {
  __shared__ _Float16 wlds[2][64 * 32];
  __shared__ _Float16 alds[4][16 * 64];
  const int tid = threadIdx.x, lane = tid & 31, wid = tid >> 5;

  for (int idx = tid; idx < 64 * 32; idx += 128) {
    wlds[0][idx] = (_Float16)Wl[idx];
    wlds[1][idx] = (_Float16)Wr[idx];
  }
  __syncthreads();

  v16h Bf[2][2][2];
#pragma unroll
  for (int m = 0; m < 2; ++m)
#pragma unroll
    for (int ks = 0; ks < 2; ++ks)
#pragma unroll
      for (int nh = 0; nh < 2; ++nh)
        Bf[m][ks][nh] = build_b(wlds[m], 32, lane, ks * 32, nh);

  const int tiles = (N + 15) >> 4;
  const int tile  = blockIdx.x * 4 + wid;
  if (tile >= tiles) return;
  const int row0 = tile << 4;
  const bool fullTile = (row0 + 16 <= N);  // wave-uniform

  _Float16* at = alds[wid];
  for (int t = lane; t < 16 * 16; t += 32) {  // 16 rows x 16 float4
    const int r = t >> 4, c4 = (t & 15) << 2;
    const int row = row0 + r;
    float4 v = make_float4(0.f, 0.f, 0.f, 0.f);
    if (row < N) v = *(const float4*)(x + (size_t)row * 64 + c4);
    at[r * 64 + c4 + 0] = (_Float16)v.x;
    at[r * 64 + c4 + 1] = (_Float16)v.y;
    at[r * 64 + c4 + 2] = (_Float16)v.z;
    at[r * 64 + c4 + 3] = (_Float16)v.w;
  }

  const v16h A0 = build_a(at, 64, lane, 0);
  const v16h A1 = build_a(at, 64, lane, 32);

  v8f acc[2][2];
#pragma unroll
  for (int m = 0; m < 2; ++m)
#pragma unroll
    for (int nh = 0; nh < 2; ++nh) {
      v8f c = {};
      c = wmma_f16(A0, Bf[m][0][nh], c);
      c = wmma_f16(A1, Bf[m][1][nh], c);
      acc[m][nh] = c;
    }

  const int rb = (lane & 16) ? 8 : 0;  // row sub-base from lane half
#pragma unroll
  for (int m = 0; m < 2; ++m) {
    float* out = m ? xr : xl;
    const float* bias = m ? br : bl;
#pragma unroll
    for (int nh = 0; nh < 2; ++nh) {
      const int col = nh * 16 + (lane & 15);
      const float bv = bias[col];
      float* outc = out + (size_t)(row0 + rb) * 32 + col;
      if (fullTile) {
#pragma unroll
        for (int r = 0; r < 8; ++r) outc[(size_t)r * 32] = acc[m][nh][r] + bv;
      } else {
#pragma unroll
        for (int r = 0; r < 8; ++r)
          if (row0 + rb + r < N) outc[(size_t)r * 32] = acc[m][nh][r] + bv;
      }
    }
  }
}

// ------------------------------------- logits + segment max (wave per edge)
__global__ void edge_logits_kernel(const float* __restrict__ xl,
                                   const float* __restrict__ xr,
                                   const float* __restrict__ We,
                                   const float* __restrict__ att,
                                   const int* __restrict__ ep,
                                   const int* __restrict__ es,
                                   const int* __restrict__ ev,
                                   long long E, long long Etot, float meanEA,
                                   float* __restrict__ logits,
                                   unsigned* __restrict__ menc) {
  const int lane = threadIdx.x & 31;
  long long w  = (long long)blockIdx.x * (blockDim.x >> 5) + (threadIdx.x >> 5);
  long long nw = (long long)gridDim.x * (blockDim.x >> 5);
  const float Wec = We[lane], attc = att[lane];
  for (long long e = w; e < Etot; e += nw) {
    int s, d; float ea;
    edge_fetch(e, ep, es, ev, E, meanEA, &s, &d, &ea);
    float v = xl[(size_t)s * 32 + lane] + xr[(size_t)d * 32 + lane] + ea * Wec;
    v = (v > 0.f) ? v : 0.2f * v;  // leaky_relu(0.2)
    float t = v * attc;
#pragma unroll
    for (int o = 16; o > 0; o >>= 1) t += __shfl_xor(t, o, 32);
    if (lane == 0) {
      logits[e] = t;
      atomicMax(&menc[d], encf(t));
    }
  }
}

// --------------------------------------- ez = exp(logit - m[dst]); denom sum
__global__ void edge_exp_kernel(const int* __restrict__ ep,
                                const int* __restrict__ es,
                                const int* __restrict__ ev,
                                long long E, long long Etot, float meanEA,
                                const unsigned* __restrict__ menc,
                                float* __restrict__ elog,  // in: logits, out: ez
                                float* __restrict__ denom) {
  long long e = (long long)blockIdx.x * blockDim.x + threadIdx.x;
  if (e >= Etot) return;
  int s, d; float ea;
  edge_fetch(e, ep, es, ev, E, meanEA, &s, &d, &ea);
  const float z = __expf(elog[e] - decf(menc[d]));
  elog[e] = z;
  atomicAdd(&denom[d], z);
}

// --------------------------------- h[dst] += alpha * xl[src] (wave per edge)
__global__ void edge_agg_kernel(const float* __restrict__ xl,
                                const int* __restrict__ ep,
                                const int* __restrict__ es,
                                const int* __restrict__ ev,
                                long long E, long long Etot, float meanEA,
                                const float* __restrict__ elog,
                                const float* __restrict__ denom,
                                float* __restrict__ h) {
  const int lane = threadIdx.x & 31;
  long long w  = (long long)blockIdx.x * (blockDim.x >> 5) + (threadIdx.x >> 5);
  long long nw = (long long)gridDim.x * (blockDim.x >> 5);
  for (long long e = w; e < Etot; e += nw) {
    int s, d; float ea;
    edge_fetch(e, ep, es, ev, E, meanEA, &s, &d, &ea);
    const float alpha = elog[e] / denom[d];
    atomicAdd(&h[(size_t)d * 32 + lane], alpha * xl[(size_t)s * 32 + lane]);
  }
}

// ------------- Y = [tanh](Xin @ W + b), K=32, ncols<=32, optional in: tanh(x+bg)
__global__ __launch_bounds__(128) void mlp_gemm_kernel(
    const float* __restrict__ X, const float* __restrict__ W,
    const float* __restrict__ b, const float* __restrict__ bgvec,
    float* __restrict__ Y, int N, int ncols, int outTanh) {
  __shared__ _Float16 wlds[32 * 32];
  __shared__ _Float16 alds[4][16 * 32];
  const int tid = threadIdx.x, lane = tid & 31, wid = tid >> 5;

  for (int idx = tid; idx < 32 * 32; idx += 128) {
    const int c = idx & 31;
    wlds[idx] = (c < ncols) ? (_Float16)W[(idx >> 5) * ncols + c] : (_Float16)0.f;
  }
  __syncthreads();

  const v16h B0 = build_b(wlds, 32, lane, 0, 0);
  const v16h B1 = build_b(wlds, 32, lane, 0, 1);

  const int tiles = (N + 15) >> 4;
  const int tile  = blockIdx.x * 4 + wid;
  if (tile >= tiles) return;
  const int row0 = tile << 4;
  const bool fullTile = (row0 + 16 <= N);  // wave-uniform

  _Float16* at = alds[wid];
  for (int t = lane; t < 16 * 8; t += 32) {  // 16 rows x 8 float4
    const int r = t >> 3, c4 = (t & 7) << 2;
    const int row = row0 + r;
    float4 v = make_float4(0.f, 0.f, 0.f, 0.f);
    if (row < N) v = *(const float4*)(X + (size_t)row * 32 + c4);
    if (bgvec) {
      v.x = tanhf(v.x + bgvec[c4 + 0]);
      v.y = tanhf(v.y + bgvec[c4 + 1]);
      v.z = tanhf(v.z + bgvec[c4 + 2]);
      v.w = tanhf(v.w + bgvec[c4 + 3]);
    }
    at[r * 32 + c4 + 0] = (_Float16)v.x;
    at[r * 32 + c4 + 1] = (_Float16)v.y;
    at[r * 32 + c4 + 2] = (_Float16)v.z;
    at[r * 32 + c4 + 3] = (_Float16)v.w;
  }

  const v16h A = build_a(at, 32, lane, 0);
  v8f c0 = {}, c1 = {};
  c0 = wmma_f16(A, B0, c0);
  c1 = wmma_f16(A, B1, c1);

  const int rb = (lane & 16) ? 8 : 0;
#pragma unroll
  for (int nh = 0; nh < 2; ++nh) {
    const int col = nh * 16 + (lane & 15);
    if (col >= ncols) continue;
    const float bv = b[col];
    const v8f acc = nh ? c1 : c0;
    float* outc = Y + (size_t)(row0 + rb) * 32 + col;
    if (fullTile) {
#pragma unroll
      for (int r = 0; r < 8; ++r) {
        float v = acc[r] + bv;
        if (outTanh) v = tanhf(v);
        outc[(size_t)r * 32] = v;
      }
    } else {
#pragma unroll
      for (int r = 0; r < 8; ++r) {
        float v = acc[r] + bv;
        if (outTanh) v = tanhf(v);
        if (row0 + rb + r < N) outc[(size_t)r * 32] = v;
      }
    }
  }
}

// --------------------------------------------- out = Y3[:, :H2] @ W4 + b4
__global__ __launch_bounds__(256) void classifier_kernel(
    const float* __restrict__ Y3, const float* __restrict__ W4,
    const float* __restrict__ b4, float* __restrict__ out, int N, int H2) {
  __shared__ float w4s[64];  // up to 32x2
  __shared__ float b4s[2];
  if (threadIdx.x < 2 * H2) w4s[threadIdx.x] = W4[threadIdx.x];
  if (threadIdx.x < 2) b4s[threadIdx.x] = b4[threadIdx.x];
  __syncthreads();

  const int n = blockIdx.x * blockDim.x + threadIdx.x;
  if (n >= N) return;
  float a0 = b4s[0], a1 = b4s[1];
  for (int k = 0; k < H2; ++k) {
    const float v = Y3[(size_t)n * 32 + k];
    a0 += v * w4s[k * 2 + 0];
    a1 += v * w4s[k * 2 + 1];
  }
  out[(size_t)n * 2 + 0] = a0;
  out[(size_t)n * 2 + 1] = a1;
}

// ------------------------------------------------------------------ launcher
extern "C" void kernel_launch(void* const* d_in, const int* in_sizes, int n_in,
                              void* d_out, int out_size, void* d_ws, size_t ws_size,
                              hipStream_t stream) {
  const float* x  = (const float*)d_in[0];
  const int*   ep = (const int*)d_in[1];
  const int*   es = (const int*)d_in[2];
  const int*   ev = (const int*)d_in[3];
  const float* Wl = (const float*)d_in[4];  const float* bl = (const float*)d_in[5];
  const float* Wr = (const float*)d_in[6];  const float* br = (const float*)d_in[7];
  const float* We = (const float*)d_in[8];  const float* att = (const float*)d_in[9];
  const float* bg = (const float*)d_in[10];
  const float* W1 = (const float*)d_in[11]; const float* b1 = (const float*)d_in[12];
  const float* W2 = (const float*)d_in[13]; const float* b2 = (const float*)d_in[14];
  const float* W3 = (const float*)d_in[15]; const float* b3 = (const float*)d_in[16];
  const float* W4 = (const float*)d_in[17]; const float* b4 = (const float*)d_in[18];

  const int       N    = in_sizes[0] / 64;
  const long long E    = in_sizes[1] / 2;
  const long long Etot = 3 * E + N;
  const int       H2   = in_sizes[16];      // 17
  const float     meanEA = 2.0f;            // mean of equal thirds {1,2,3}

  // workspace layout (floats), regions 16B-aligned
  float*    xl    = (float*)d_ws;
  float*    xr    = xl + (size_t)N * 32;
  unsigned* menc  = (unsigned*)(xr + (size_t)N * 32);
  float*    denom = (float*)(menc + N);
  float*    elog  = denom + N;
  const long long EtotPad = (Etot + 63) & ~63LL;
  float*    hbuf  = elog + EtotPad;
  float*    h2buf = hbuf + (size_t)N * 32;

  const int tiles = (N + 15) >> 4;
  const unsigned gemmBlocks = (unsigned)((tiles + 3) / 4);

  init_kernel<<<(unsigned)(((long long)N * 32 + 255) / 256), 256, 0, stream>>>(
      menc, denom, hbuf, N);

  node_transform_kernel<<<gemmBlocks, 128, 0, stream>>>(x, Wl, bl, Wr, br, xl, xr, N);

  edge_logits_kernel<<<2048, 256, 0, stream>>>(xl, xr, We, att, ep, es, ev, E, Etot,
                                               meanEA, elog, menc);

  edge_exp_kernel<<<(unsigned)((Etot + 255) / 256), 256, 0, stream>>>(
      ep, es, ev, E, Etot, meanEA, menc, elog, denom);

  edge_agg_kernel<<<2048, 256, 0, stream>>>(xl, ep, es, ev, E, Etot, meanEA, elog,
                                            denom, hbuf);

  // X0 = tanh(h + bg); T1 = tanh(X0@W1+b1)
  mlp_gemm_kernel<<<gemmBlocks, 128, 0, stream>>>(hbuf, W1, b1, bg, h2buf, N, 32, 1);
  // X1 = T1@W2 + b2
  mlp_gemm_kernel<<<gemmBlocks, 128, 0, stream>>>(h2buf, W2, b2, nullptr, hbuf, N, 32, 0);
  // T2 = tanh(X1@W3 + b3)   (17 valid cols)
  mlp_gemm_kernel<<<gemmBlocks, 128, 0, stream>>>(hbuf, W3, b3, nullptr, h2buf, N, H2, 1);
  // out = T2 @ W4 + b4
  classifier_kernel<<<(unsigned)((N + 255) / 256), 256, 0, stream>>>(
      h2buf, W4, b4, (float*)d_out, N, H2);
}
